// SelfAttention_549755814307
// MI455X (gfx1250) — matrix operand order repaired and checked by reading, not compiled, not run on hardware
//
#include <hip/hip_runtime.h>
#include <hip/hip_bf16.h>

typedef __attribute__((ext_vector_type(16))) _Float16 v16h;
typedef __attribute__((ext_vector_type(8)))  _Float16 v8h;
typedef __attribute__((ext_vector_type(8)))  float    v8f;
typedef __attribute__((ext_vector_type(2)))  float    v2f;

// ---------------------------------------------------------------------------
// WMMA fragment loaders for v_wmma_f32_16x16x32_f16 (gfx1250, wave32).
// Per ISA 7.12.2:
//   A (16x32, MxK): lane l<16 -> row l, k = {koff..koff+7, koff+16..koff+23},
//                   koff = 0;  lane l>=16 -> row l-16, koff = 8.
//   B (32x16, KxN): lane l<16 -> col l, k = {koff..koff+7, koff+16..koff+23},
//                   koff = 0;  lane l>=16 -> col l-16, koff = 8.
// `base` points at element [row0/col0][k0]; `ld` is the row/col stride in
// halves. All strides chosen 16B-aligned.
// ---------------------------------------------------------------------------
__device__ __forceinline__ v16h frag_a_f16(const _Float16* __restrict__ base, int ld) {
  const int l    = threadIdx.x & 31;
  const int row  = l & 15;
  const int koff = (l < 16) ? 0 : 8;
  const _Float16* p = base + row * ld + koff;
  v8h lo = *(const v8h*)(p);        // k = koff .. koff+7
  v8h hi = *(const v8h*)(p + 16);   // k = koff+16 .. koff+23
  v16h r;
#pragma unroll
  for (int i = 0; i < 8; ++i) { r[i] = lo[i]; r[i + 8] = hi[i]; }
  return r;
}

__device__ __forceinline__ v16h frag_b_f16(const _Float16* __restrict__ base, int ld) {
  const int l    = threadIdx.x & 31;
  const int col  = l & 15;
  const int koff = (l < 16) ? 0 : 8;
  const _Float16* p = base + col * ld + koff;
  v8h lo = *(const v8h*)(p);        // k = koff .. koff+7
  v8h hi = *(const v8h*)(p + 16);   // k = koff+16 .. koff+23
  v16h r;
#pragma unroll
  for (int i = 0; i < 8; ++i) { r[i] = lo[i]; r[i + 8] = hi[i]; }
  return r;
}

#define WMMA_F16(a, b, c) \
  __builtin_amdgcn_wmma_f32_16x16x32_f16(false, (a), false, (b), (short)0, (c), false, false)

// ---------------------------------------------------------------------------
// Kernel 0: one-shot f32 -> f16 conversion of the three 256x256 weights.
// Removes all per-block weight conversion VALU work from the projection
// kernel and halves the weight-fetch bytes (f16 copies stay L2-resident).
// ---------------------------------------------------------------------------
__global__ __launch_bounds__(256) void convert_w_kernel(
    const float* __restrict__ wq, const float* __restrict__ wk,
    const float* __restrict__ wv, _Float16* __restrict__ wh)
{
  constexpr int WSZ = 256 * 256;
  const int idx = blockIdx.x * 256 + threadIdx.x;   // grid covers 3*WSZ/2
  const float* srcs[3] = {wq, wk, wv};
  const int mat = idx / (WSZ / 2);                  // 2 elements per thread
  const int e   = (idx % (WSZ / 2)) * 2;
  v2f v = *(const v2f*)(srcs[mat] + e);
  _Float16* dst = wh + (size_t)mat * WSZ + e;
  dst[0] = (_Float16)v[0];
  dst[1] = (_Float16)v[1];
}

// ---------------------------------------------------------------------------
// Kernel 1: Q/K/V projections.  q[b,n,o] = sum_c wq[o,c]*x[b,c,n] + bq[o]
// Block = (batch, 64-token tile), 256 threads = 8 waves.
// Wave w: token subtile (w>>1)*16, output half (w&1)*128.
// Outputs token-major f16 [B, N, 256].
// ---------------------------------------------------------------------------
__global__ __launch_bounds__(256) void qkv_proj_kernel(
    const float* __restrict__ x,
    const _Float16* __restrict__ wh,     // [3][256*256] f16 (q,k,v)
    const float* __restrict__ bq, const float* __restrict__ bk,
    const float* __restrict__ bv,
    _Float16* __restrict__ qt, _Float16* __restrict__ kt, _Float16* __restrict__ vt)
{
  constexpr int C = 256, N = 4096;
  constexpr int LDX = 264;                 // 256 + 8 pad halves (528B, 16B aligned)
  __shared__ _Float16 xs[64 * LDX];        // 33 KB: x tile, token-major f16

  const int b   = blockIdx.y;
  const int t0  = blockIdx.x * 64;
  const int tid = threadIdx.x;

  // Cooperative transpose load: x[b, c, t0+n] (f32) -> xs[n][c] (f16)
  const float* xb = x + (size_t)b * C * N + t0;
  for (int idx = tid; idx < 64 * C; idx += 256) {
    const int c = idx >> 6, n = idx & 63;  // coalesced over n within each c row
    xs[n * LDX + c] = (_Float16)xb[(size_t)c * N + n];
  }
  __syncthreads();

  const int w  = tid >> 5;
  const int ts = (w >> 1) * 16;            // token subtile base (0..48)
  const int h  = (w & 1) * 128;            // output-channel half base
  const int l  = tid & 31;

  // Hoist A fragments: 16 tokens x 256 channels (8 k-chunks)
  v16h afrag[8];
#pragma unroll
  for (int kc = 0; kc < 8; ++kc)
    afrag[kc] = frag_a_f16(xs + ts * LDX + kc * 32, LDX);

  const float* Bs[3] = {bq, bk, bv};
  _Float16*    Os[3] = {qt, kt, vt};

#pragma unroll
  for (int pj = 0; pj < 3; ++pj) {
    const _Float16* W  = wh + (size_t)pj * C * C;
    const float*    Bv = Bs[pj];
    _Float16*       O  = Os[pj] + ((size_t)b * N + t0 + ts) * C;
#pragma unroll
    for (int ot = 0; ot < 8; ++ot) {
      const int o0 = h + ot * 16;
      const float bias = Bv[o0 + (l & 15)];     // C/D col = lane%16 for all VGPRs
      v8f acc;
#pragma unroll
      for (int i = 0; i < 8; ++i) acc[i] = bias;
#pragma unroll
      for (int kc = 0; kc < 8; ++kc) {
        v16h bfrag = frag_b_f16(W + (size_t)o0 * C + kc * 32, C);
        acc = WMMA_F16(afrag[kc], bfrag, acc);
      }
      // Store: VGPR r, lanes 0-15 -> token ts+r; lanes 16-31 -> token ts+8+r.
      const int oc    = o0 + (l & 15);
      const int rbase = (l < 16) ? 0 : 8;
#pragma unroll
      for (int r = 0; r < 8; ++r)
        O[(size_t)(rbase + r) * C + oc] = (_Float16)acc[r];
    }
  }
}

// ---------------------------------------------------------------------------
// Kernel 2: flash-attention.  Block = (batch, 128 queries), 8 waves x 16 rows.
// Key loop in 32-token tiles; S=QK^T via WMMA, online softmax, O += P V^T.
// Writes O (pre-gamma, post-1/l) token-major f16 [B, N, 256].
// ---------------------------------------------------------------------------
__global__ __launch_bounds__(256) void attn_kernel(
    const _Float16* __restrict__ qt, const _Float16* __restrict__ kt,
    const _Float16* __restrict__ vt, _Float16* __restrict__ ot)
{
  constexpr int C = 256, N = 4096;
  constexpr int LDK = 264;  // ks row stride (halves)
  constexpr int LDV = 40;   // vs row stride: 32 tokens + 8 pad
  constexpr int LDP = 40;   // ps row stride: 32 cols + 8 pad
  __shared__ _Float16 ks[32 * LDK];      // K tile, token-major       (16.5 KB)
  __shared__ _Float16 vs[C * LDV];       // V tile, channel-major     (20 KB)
  __shared__ _Float16 ps[8 * 16 * LDP];  // per-wave P scratch        (10 KB)

  const int b   = blockIdx.y;
  const int q0  = blockIdx.x * 128;
  const int tid = threadIdx.x;
  const int w   = tid >> 5;
  const int l   = tid & 31;
  const int qw  = q0 + w * 16;           // this wave's first query token

  const _Float16* qb = qt + (size_t)b * N * C;
  const _Float16* kp = kt + (size_t)b * N * C;
  const _Float16* vp = vt + (size_t)b * N * C;

  // Hoisted Q fragments: 16 query rows x full 256 channels
  v16h aq[8];
#pragma unroll
  for (int kc = 0; kc < 8; ++kc)
    aq[kc] = frag_a_f16(qb + (size_t)qw * C + kc * 32, C);

  v8f oacc[16];
#pragma unroll
  for (int ct = 0; ct < 16; ++ct)
#pragma unroll
    for (int i = 0; i < 8; ++i) oacc[ct][i] = 0.f;

  float rm[8], rl[8];                    // per-lane: stats of row (r + lanehalf*8)
#pragma unroll
  for (int r = 0; r < 8; ++r) { rm[r] = -1e30f; rl[r] = 0.f; }

  _Float16* psw = ps + w * 16 * LDP;

  for (int m0 = 0; m0 < N; m0 += 32) {
    __syncthreads();
    // Cooperative: K tile -> ks[m][c]; V tile transposed -> vs[c][m]
    for (int idx = tid; idx < 32 * C; idx += 256) {
      const int m = idx >> 8, c = idx & 255;
      ks[m * LDK + c] = kp[(size_t)(m0 + m) * C + c];
      vs[c * LDV + m] = vp[(size_t)(m0 + m) * C + c];
    }
    __syncthreads();

    // S (16x32) = Q_tile * K_tile^T : two 16x16 WMMA accumulators
    v8f s0, s1;
#pragma unroll
    for (int i = 0; i < 8; ++i) { s0[i] = 0.f; s1[i] = 0.f; }
#pragma unroll
    for (int kc = 0; kc < 8; ++kc) {
      v16h b0 = frag_b_f16(ks + kc * 32, LDK);            // keys m0..m0+15
      v16h b1 = frag_b_f16(ks + 16 * LDK + kc * 32, LDK); // keys m0+16..m0+31
      s0 = WMMA_F16(aq[kc], b0, s0);
      s1 = WMMA_F16(aq[kc], b1, s1);
    }

    // Online softmax per row (row = r + 8*(l>=16); its 16 cols live across
    // the 16 lanes of the half-group -> xor-butterfly over masks 1,2,4,8).
#pragma unroll
    for (int r = 0; r < 8; ++r) {
      const float a0 = s0[r], a1 = s1[r];
      float mx = fmaxf(a0, a1);
#pragma unroll
      for (int msk = 1; msk < 16; msk <<= 1)
        mx = fmaxf(mx, __shfl_xor(mx, msk, 32));
      const float mnew = fmaxf(rm[r], mx);
      const float corr = __expf(rm[r] - mnew);
      const float p0   = __expf(a0 - mnew);
      const float p1   = __expf(a1 - mnew);
      float sum = p0 + p1;
#pragma unroll
      for (int msk = 1; msk < 16; msk <<= 1)
        sum += __shfl_xor(sum, msk, 32);
      rl[r] = rl[r] * corr + sum;
      rm[r] = mnew;
#pragma unroll
      for (int ct = 0; ct < 16; ++ct) oacc[ct][r] *= corr;   // rescale O row
      const int prow = r + ((l < 16) ? 0 : 8);
      psw[prow * LDP + (l & 15)]      = (_Float16)p0;
      psw[prow * LDP + 16 + (l & 15)] = (_Float16)p1;
    }

    // O += P (16x32) * V^T (32x256); B[k=m][n=c] = vs[c][m] (contiguous)
    v16h pf = frag_a_f16(psw, LDP);
#pragma unroll
    for (int ct = 0; ct < 16; ++ct) {
      v16h bf = frag_b_f16(vs + (ct * 16) * LDV, LDV);
      oacc[ct] = WMMA_F16(pf, bf, oacc[ct]);
    }
  }

  // Store O / l, token-major f16
  _Float16* ob = ot + ((size_t)b * N + qw) * C;
  const int rbase = (l < 16) ? 0 : 8;
#pragma unroll
  for (int r = 0; r < 8; ++r) {
    const float inv = 1.0f / rl[r];
#pragma unroll
    for (int ct = 0; ct < 16; ++ct)
      ob[(size_t)(rbase + r) * C + ct * 16 + (l & 15)] = (_Float16)(oacc[ct][r] * inv);
  }
}

// ---------------------------------------------------------------------------
// Kernel 3: out[b,c,n] = gamma * O[b,n,c] + x[b,c,n]   (coalesced on out/x)
// ---------------------------------------------------------------------------
__global__ __launch_bounds__(256) void epilogue_kernel(
    const float* __restrict__ x, const _Float16* __restrict__ ot,
    const float* __restrict__ gamma, float* __restrict__ out)
{
  constexpr int C = 256, N = 4096;
  const size_t idx = (size_t)blockIdx.x * 256 + threadIdx.x;
  const size_t total = (size_t)8 * C * N;
  if (idx >= total) return;
  const size_t n = idx % N;
  const size_t c = (idx / N) % C;
  const size_t b = idx / ((size_t)N * C);
  out[idx] = gamma[0] * (float)ot[((size_t)b * N + n) * C + c] + x[idx];
}

// ---------------------------------------------------------------------------
extern "C" void kernel_launch(void* const* d_in, const int* in_sizes, int n_in,
                              void* d_out, int out_size, void* d_ws, size_t ws_size,
                              hipStream_t stream) {
  constexpr int B = 8, C = 256, N = 4096;
  const float* x     = (const float*)d_in[0];
  const float* wq    = (const float*)d_in[1];
  const float* bq    = (const float*)d_in[2];
  const float* wk    = (const float*)d_in[3];
  const float* bk    = (const float*)d_in[4];
  const float* wv    = (const float*)d_in[5];
  const float* bv    = (const float*)d_in[6];
  const float* gamma = (const float*)d_in[7];
  float* out = (float*)d_out;

  const size_t elems = (size_t)B * N * C;      // 8.39M elems per f16 tensor
  _Float16* qt = (_Float16*)d_ws;              // 16 MB each
  _Float16* kt = qt + elems;
  _Float16* vt = kt + elems;
  _Float16* ot = vt + elems;
  _Float16* wh = ot + elems;                   // [3][65536] f16 weights (384 KB)

  // Kernel 0: weight conversion (3*65536 elems, 2 per thread)
  convert_w_kernel<<<(3 * C * C / 2) / 256, 256, 0, stream>>>(wq, wk, wv, wh);

  dim3 g1(N / 64, B);
  qkv_proj_kernel<<<g1, 256, 0, stream>>>(x, wh, bq, bk, bv, qt, kt, vt);

  dim3 g2(N / 128, B);
  attn_kernel<<<g2, 256, 0, stream>>>(qt, kt, vt, ot);

  const size_t total = (size_t)B * C * N;
  epilogue_kernel<<<(unsigned)((total + 255) / 256), 256, 0, stream>>>(x, ot, gamma, out);
}
